// HATGPT_37263136260732
// MI455X (gfx1250) — matrix operand order repaired and checked by reading, not compiled, not run on hardware
//
#include <hip/hip_runtime.h>
#include <hip/hip_bf16.h>
#include <math.h>

// ---------------- model dims (fixed by reference) ----------------
#define D_      1024
#define H_      16
#define HD_     64
#define DFF_    4096
#define LAYERS_ 4
#define SEQ_    2048
#define B_      2
#define NTOK_   (B_ * SEQ_)   // 4096
#define VOCAB_  32000

// ---------------- types ----------------
typedef __bf16 bf16_t;
typedef __attribute__((ext_vector_type(16))) __bf16 v16bf;
typedef __attribute__((ext_vector_type(8)))  __bf16 v8bf;
typedef __attribute__((ext_vector_type(8)))  float  v8f;

__device__ __forceinline__ bf16_t f2bf(float f) {
  unsigned u = __builtin_bit_cast(unsigned, f);
  unsigned r = u + 0x7fffu + ((u >> 16) & 1u);   // round-to-nearest-even
  unsigned short hs = (unsigned short)(r >> 16);
  return __builtin_bit_cast(bf16_t, hs);
}

union FragU { v16bf v; v8bf h[2]; };

// A-matrix 16x32 bf16 fragment (cdna5 CDNA5 ISA):
// lanes 0-15: row = lane, K = {0..7, 16..23}; lanes 16-31: row = lane-16, K = {8..15, 24..31}
__device__ __forceinline__ v16bf load_a_frag(const bf16_t* tile, int ld) {
  int lane = threadIdx.x & 31;
  const bf16_t* p = tile + (size_t)(lane & 15) * ld + ((lane >> 4) << 3);
  FragU f;
  f.h[0] = *(const v8bf*)p;
  f.h[1] = *(const v8bf*)(p + 16);
  return f.v;
}

// B-matrix 32x16 bf16 fragment:
// lanes 0-15: col(N) = lane, K = {0..7, 16..23}; lanes 16-31: col = lane-16, K = {8..15, 24..31}
__device__ __forceinline__ v16bf load_b_frag(const bf16_t* tile, int ld) {
  int lane = threadIdx.x & 31;
  const bf16_t* p = tile + (size_t)(lane & 15) * ld + ((lane >> 4) << 3);
  FragU f;
  f.h[0] = *(const v8bf*)p;
  f.h[1] = *(const v8bf*)(p + 16);
  return f.v;
}

__device__ __forceinline__ v8f wmma_bf(v16bf a, v16bf b, v8f c) {
  return __builtin_amdgcn_wmma_f32_16x16x32_bf16(false, a, false, b, (short)0, c,
                                                 false, false);
}

// reductions across a 16-lane half (wave32)
__device__ __forceinline__ float hmax16(float x) {
  x = fmaxf(x, __shfl_xor(x, 1, 32));
  x = fmaxf(x, __shfl_xor(x, 2, 32));
  x = fmaxf(x, __shfl_xor(x, 4, 32));
  x = fmaxf(x, __shfl_xor(x, 8, 32));
  return x;
}
__device__ __forceinline__ float hsum16(float x) {
  x += __shfl_xor(x, 1, 32);
  x += __shfl_xor(x, 2, 32);
  x += __shfl_xor(x, 4, 32);
  x += __shfl_xor(x, 8, 32);
  return x;
}

// ---------------- embedding: h[t,:] = tok_emb[x[t],:] + pos_emb[s,:] ----------------
__launch_bounds__(256)
__global__ void embed_kernel(const int* __restrict__ x, const float* __restrict__ tok,
                             const float* __restrict__ pos, float* __restrict__ hbuf) {
  int t = blockIdx.x;
  int s = t & (SEQ_ - 1);
  int tokid = x[t];
  const float* tr = tok + (size_t)tokid * D_;
  const float* pr = pos + (size_t)s * D_;
  float* hr = hbuf + (size_t)t * D_;
  for (int d = threadIdx.x; d < D_; d += 256) hr[d] = tr[d] + pr[d];
}

// ---------------- LayerNorm -> bf16 ----------------
__launch_bounds__(256)
__global__ void ln_bf16_kernel(const float* __restrict__ x, const float* __restrict__ g,
                               const float* __restrict__ bta, bf16_t* __restrict__ y) {
  int t = blockIdx.x;
  const float* row = x + (size_t)t * D_;
  float v[4];
  float s = 0.f, s2 = 0.f;
#pragma unroll
  for (int i = 0; i < 4; ++i) {
    v[i] = row[threadIdx.x + i * 256];
    s += v[i];
    s2 += v[i] * v[i];
  }
#pragma unroll
  for (int off = 16; off >= 1; off >>= 1) {
    s  += __shfl_xor(s,  off, 32);
    s2 += __shfl_xor(s2, off, 32);
  }
  __shared__ float sh[16];
  int wid = threadIdx.x >> 5, lane = threadIdx.x & 31;
  if (lane == 0) { sh[wid] = s; sh[8 + wid] = s2; }
  __syncthreads();
  float S = 0.f, S2 = 0.f;
#pragma unroll
  for (int w = 0; w < 8; ++w) { S += sh[w]; S2 += sh[8 + w]; }
  float mu = S * (1.f / D_);
  float var = S2 * (1.f / D_) - mu * mu;
  float rstd = rsqrtf(var + 1e-5f);
  bf16_t* yr = y + (size_t)t * D_;
#pragma unroll
  for (int i = 0; i < 4; ++i) {
    int d = threadIdx.x + i * 256;
    yr[d] = f2bf((v[i] - mu) * rstd * g[d] + bta[d]);
  }
}

// ---------------- weight convert+transpose: w[KxN] fp32 -> wt[NxK] bf16 ----------------
__launch_bounds__(256)
__global__ void wtrans_kernel(const float* __restrict__ w, bf16_t* __restrict__ wt,
                              int K, int N) {
  __shared__ float tile[32][33];
  int n0 = blockIdx.x * 32, k0 = blockIdx.y * 32;
  int tx = threadIdx.x & 31, ty = threadIdx.x >> 5;
#pragma unroll
  for (int i = 0; i < 32; i += 8)
    tile[ty + i][tx] = w[(size_t)(k0 + ty + i) * N + (n0 + tx)];
  __syncthreads();
#pragma unroll
  for (int i = 0; i < 32; i += 8)
    wt[(size_t)(n0 + ty + i) * K + (k0 + tx)] = f2bf(tile[tx][ty + i]);
}

// ---------------- V pack: v[t, h*HD+hd] bf16 -> vt[((b*H+h)*HD+hd)*SEQ + s] ----------------
__launch_bounds__(256)
__global__ void vpack_kernel(const bf16_t* __restrict__ v, bf16_t* __restrict__ vt) {
  size_t idx = (size_t)blockIdx.x * 256 + threadIdx.x;  // over b,h,hd,s (s fastest)
  int s = (int)(idx & (SEQ_ - 1));
  size_t rest = idx >> 11;
  int hd = (int)(rest & (HD_ - 1));
  size_t bh = rest >> 6;
  int hh = (int)(bh & (H_ - 1));
  int b = (int)(bh >> 4);
  vt[idx] = v[(size_t)(b * SEQ_ + s) * D_ + hh * HD_ + hd];
}

// ---------------- WMMA GEMM: out = A[MxK] * Bt[NxK]^T + bias ----------------
// MODE 0: outF = acc + bias                       (logits)
// MODE 1: outF = acc + bias + resid  (fp32)       (wo / lin2 with residual, in-place ok)
// MODE 2: outB = gelu(acc + bias)    (bf16)       (lin1)
// MODE 3: outB = acc + bias          (bf16)       (q/k/v)
template <int MODE>
__launch_bounds__(256)
__global__ void gemm_bf16_kernel(const bf16_t* __restrict__ A, const bf16_t* __restrict__ Bt,
                                 const float* __restrict__ bias, const float* __restrict__ resid,
                                 float* __restrict__ outF, bf16_t* __restrict__ outB,
                                 int M, int N, int K) {
  int wid = threadIdx.x >> 5, lane = threadIdx.x & 31;
  int mbase = blockIdx.y * 128 + (wid & 3) * 32;   // 4 waves along M
  int nbase = blockIdx.x * 128 + (wid >> 2) * 64;  // 2 waves along N
  v8f acc[2][4] = {};
  const bf16_t* Arow = A + (size_t)mbase * K;
  const bf16_t* Brow = Bt + (size_t)nbase * K;
  for (int kk = 0; kk < K; kk += 32) {
    v16bf a0 = load_a_frag(Arow + kk, K);
    v16bf a1 = load_a_frag(Arow + (size_t)16 * K + kk, K);
#pragma unroll
    for (int j = 0; j < 4; ++j) {
      v16bf bf = load_b_frag(Brow + (size_t)(16 * j) * K + kk, K);
      acc[0][j] = wmma_bf(a0, bf, acc[0][j]);
      acc[1][j] = wmma_bf(a1, bf, acc[1][j]);
    }
  }
  int hi = lane >> 4, col = lane & 15;
#pragma unroll
  for (int j = 0; j < 4; ++j) {
    int n = nbase + 16 * j + col;
    float bv = bias[n];
#pragma unroll
    for (int i = 0; i < 2; ++i) {
#pragma unroll
      for (int r = 0; r < 8; ++r) {
        int m = mbase + 16 * i + r + hi * 8;
        size_t idx = (size_t)m * N + n;
        float vv = acc[i][j][r] + bv;
        if (MODE == 0) {
          outF[idx] = vv;
        } else if (MODE == 1) {
          outF[idx] = vv + resid[idx];
        } else if (MODE == 2) {
          outB[idx] = f2bf(0.5f * vv * (1.f + erff(vv * 0.70710678118654752f)));
        } else {
          outB[idx] = f2bf(vv);
        }
      }
    }
  }
}

// ---------------- flash attention: 1 wave per 16-query tile ----------------
// q,k: bf16 [t, h*HD+hd] ; vt: bf16 [b,h,hd,s] ; outp: bf16 [t, h*HD+hd]
__launch_bounds__(256)
__global__ void attn_kernel(const bf16_t* __restrict__ q, const bf16_t* __restrict__ k,
                            const bf16_t* __restrict__ vt, bf16_t* __restrict__ outp) {
  __shared__ __align__(16) bf16_t ptile[8][16 * 32];
  int wid = threadIdx.x >> 5, lane = threadIdx.x & 31;
  int tileId = blockIdx.x * 8 + wid;
  int qt = tileId & (SEQ_ / 16 - 1);   // 128 tiles per (b,h)
  int bh = tileId >> 7;
  int b = bh >> 4, hh = bh & (H_ - 1);
  int qbase = qt * 16;
  int hi = lane >> 4, col = lane & 15;

  const bf16_t* qp = q + (size_t)(b * SEQ_ + qbase) * D_ + hh * HD_;
  v16bf qf0 = load_a_frag(qp, D_);
  v16bf qf1 = load_a_frag(qp + 32, D_);

  const bf16_t* kp = k + (size_t)(b * SEQ_) * D_ + hh * HD_;
  const bf16_t* vp = vt + (size_t)(b * H_ + hh) * HD_ * SEQ_;

  v8f o0 = {}, o1 = {}, o2 = {}, o3 = {};
  float mrow[8], lrow[8];
#pragma unroll
  for (int r = 0; r < 8; ++r) { mrow[r] = -3.0e38f; lrow[r] = 0.f; }

  int cmax = (qbase + 15) >> 5;
  for (int c = 0; c <= cmax; ++c) {
    int kc = c * 32;
    // S[16x32] = Q[16x64] * K^T[64x32], in two 16-col fragments
    v8f s0 = {}, s1 = {};
    v16bf kf;
    kf = load_b_frag(kp + (size_t)kc * D_,            D_); s0 = wmma_bf(qf0, kf, s0);
    kf = load_b_frag(kp + (size_t)kc * D_ + 32,       D_); s0 = wmma_bf(qf1, kf, s0);
    kf = load_b_frag(kp + (size_t)(kc + 16) * D_,     D_); s1 = wmma_bf(qf0, kf, s1);
    kf = load_b_frag(kp + (size_t)(kc + 16) * D_ + 32, D_); s1 = wmma_bf(qf1, kf, s1);

    bool domask = (c == cmax);
#pragma unroll
    for (int r = 0; r < 8; ++r) {
      int qrow = qbase + r + hi * 8;
      float sa = s0[r] * 0.125f;   // 1/sqrt(HD)
      float sb = s1[r] * 0.125f;
      if (domask) {
        if (kc + col > qrow)      sa = -1e30f;
        if (kc + 16 + col > qrow) sb = -1e30f;
      }
      float mx = hmax16(fmaxf(sa, sb));
      float mn = fmaxf(mrow[r], mx);
      float pa = __expf(sa - mn);
      float pb = __expf(sb - mn);
      float rs = hsum16(pa + pb);
      float sc = __expf(mrow[r] - mn);
      lrow[r] = lrow[r] * sc + rs;
      mrow[r] = mn;
      o0[r] *= sc; o1[r] *= sc; o2[r] *= sc; o3[r] *= sc;
      int row = r + hi * 8;
      ptile[wid][row * 32 + col]      = f2bf(pa);
      ptile[wid][row * 32 + 16 + col] = f2bf(pb);
    }
    // LDS store -> load (same wave, aliasing through memory): explicit CDNA5 wait
    asm volatile("s_wait_dscnt 0" ::: "memory");
    v16bf pf = load_a_frag(&ptile[wid][0], 32);
    v16bf vf;
    vf = load_b_frag(vp + (size_t)0  * SEQ_ + kc, SEQ_); o0 = wmma_bf(pf, vf, o0);
    vf = load_b_frag(vp + (size_t)16 * SEQ_ + kc, SEQ_); o1 = wmma_bf(pf, vf, o1);
    vf = load_b_frag(vp + (size_t)32 * SEQ_ + kc, SEQ_); o2 = wmma_bf(pf, vf, o2);
    vf = load_b_frag(vp + (size_t)48 * SEQ_ + kc, SEQ_); o3 = wmma_bf(pf, vf, o3);
  }
#pragma unroll
  for (int r = 0; r < 8; ++r) {
    float inv = 1.f / lrow[r];
    size_t base = (size_t)(b * SEQ_ + qbase + r + hi * 8) * D_ + hh * HD_ + col;
    outp[base + 0]  = f2bf(o0[r] * inv);
    outp[base + 16] = f2bf(o1[r] * inv);
    outp[base + 32] = f2bf(o2[r] * inv);
    outp[base + 48] = f2bf(o3[r] * inv);
  }
}

// ---------------- host orchestration ----------------
extern "C" void kernel_launch(void* const* d_in, const int* in_sizes, int n_in,
                              void* d_out, int out_size, void* d_ws, size_t ws_size,
                              hipStream_t stream) {
  (void)in_sizes; (void)n_in; (void)out_size; (void)ws_size;
  // input order: 0 x, 1 mask(unused), 2 tok_emb, 3 pos_emb,
  // per layer l (base 4+16l): wq.w,wq.b, wk.w,wk.b, wv.w,wv.b, wo.w,wo.b,
  //                           lin1.w,lin1.b, lin2.w,lin2.b, ln1.g,ln1.b, ln2.g,ln2.b
  // 68 ln_f.g, 69 ln_f.b, 70 out.w, 71 out.b
  const int* x = (const int*)d_in[0];
  const float* tok_emb = (const float*)d_in[2];
  const float* pos_emb = (const float*)d_in[3];

  char* ws = (char*)d_ws;
  size_t off = 0;
  auto carve = [&](size_t bytes) -> void* {
    void* p = ws + off;
    off += (bytes + 255) & ~(size_t)255;
    return p;
  };

  bf16_t *wt_q[LAYERS_], *wt_k[LAYERS_], *wt_v[LAYERS_], *wt_o[LAYERS_];
  bf16_t *wt_1[LAYERS_], *wt_2[LAYERS_];
  for (int l = 0; l < LAYERS_; ++l) {
    wt_q[l] = (bf16_t*)carve((size_t)D_ * D_ * 2);
    wt_k[l] = (bf16_t*)carve((size_t)D_ * D_ * 2);
    wt_v[l] = (bf16_t*)carve((size_t)D_ * D_ * 2);
    wt_o[l] = (bf16_t*)carve((size_t)D_ * D_ * 2);
    wt_1[l] = (bf16_t*)carve((size_t)D_ * DFF_ * 2);
    wt_2[l] = (bf16_t*)carve((size_t)D_ * DFF_ * 2);
  }
  bf16_t* wt_out = (bf16_t*)carve((size_t)D_ * VOCAB_ * 2);
  float*  hbuf   = (float*) carve((size_t)NTOK_ * D_ * 4);
  bf16_t* xn     = (bf16_t*)carve((size_t)NTOK_ * D_ * 2);
  bf16_t* qbuf   = (bf16_t*)carve((size_t)NTOK_ * D_ * 2);
  bf16_t* kbuf   = (bf16_t*)carve((size_t)NTOK_ * D_ * 2);
  bf16_t* vbuf   = (bf16_t*)carve((size_t)NTOK_ * D_ * 2);
  bf16_t* vtbuf  = (bf16_t*)carve((size_t)NTOK_ * D_ * 2);
  bf16_t* attnb  = (bf16_t*)carve((size_t)NTOK_ * D_ * 2);
  bf16_t* midb   = (bf16_t*)carve((size_t)NTOK_ * DFF_ * 2);

  dim3 blk(256);

  // one-time weight convert+transpose to bf16 [NxK]
  for (int l = 0; l < LAYERS_; ++l) {
    int base = 4 + l * 16;
    wtrans_kernel<<<dim3(D_ / 32, D_ / 32), blk, 0, stream>>>((const float*)d_in[base + 0], wt_q[l], D_, D_);
    wtrans_kernel<<<dim3(D_ / 32, D_ / 32), blk, 0, stream>>>((const float*)d_in[base + 2], wt_k[l], D_, D_);
    wtrans_kernel<<<dim3(D_ / 32, D_ / 32), blk, 0, stream>>>((const float*)d_in[base + 4], wt_v[l], D_, D_);
    wtrans_kernel<<<dim3(D_ / 32, D_ / 32), blk, 0, stream>>>((const float*)d_in[base + 6], wt_o[l], D_, D_);
    wtrans_kernel<<<dim3(DFF_ / 32, D_ / 32), blk, 0, stream>>>((const float*)d_in[base + 8], wt_1[l], D_, DFF_);
    wtrans_kernel<<<dim3(D_ / 32, DFF_ / 32), blk, 0, stream>>>((const float*)d_in[base + 10], wt_2[l], DFF_, D_);
  }
  wtrans_kernel<<<dim3(VOCAB_ / 32, D_ / 32), blk, 0, stream>>>((const float*)d_in[70], wt_out, D_, VOCAB_);

  embed_kernel<<<NTOK_, blk, 0, stream>>>(x, tok_emb, pos_emb, hbuf);

  dim3 gD(D_ / 128, NTOK_ / 128);      // N=1024 GEMMs
  dim3 gF(DFF_ / 128, NTOK_ / 128);    // N=4096 GEMM
  for (int l = 0; l < LAYERS_; ++l) {
    int base = 4 + l * 16;
    const float* bq = (const float*)d_in[base + 1];
    const float* bk = (const float*)d_in[base + 3];
    const float* bv = (const float*)d_in[base + 5];
    const float* bo = (const float*)d_in[base + 7];
    const float* b1 = (const float*)d_in[base + 9];
    const float* b2 = (const float*)d_in[base + 11];
    const float* ln1g = (const float*)d_in[base + 12];
    const float* ln1b = (const float*)d_in[base + 13];
    const float* ln2g = (const float*)d_in[base + 14];
    const float* ln2b = (const float*)d_in[base + 15];

    ln_bf16_kernel<<<NTOK_, blk, 0, stream>>>(hbuf, ln1g, ln1b, xn);
    gemm_bf16_kernel<3><<<gD, blk, 0, stream>>>(xn, wt_q[l], bq, nullptr, nullptr, qbuf, NTOK_, D_, D_);
    gemm_bf16_kernel<3><<<gD, blk, 0, stream>>>(xn, wt_k[l], bk, nullptr, nullptr, kbuf, NTOK_, D_, D_);
    gemm_bf16_kernel<3><<<gD, blk, 0, stream>>>(xn, wt_v[l], bv, nullptr, nullptr, vbuf, NTOK_, D_, D_);
    vpack_kernel<<<(NTOK_ * D_) / 256, blk, 0, stream>>>(vbuf, vtbuf);
    attn_kernel<<<(B_ * H_ * (SEQ_ / 16)) / 8, blk, 0, stream>>>(qbuf, kbuf, vtbuf, attnb);
    gemm_bf16_kernel<1><<<gD, blk, 0, stream>>>(attnb, wt_o[l], bo, hbuf, hbuf, nullptr, NTOK_, D_, D_);

    ln_bf16_kernel<<<NTOK_, blk, 0, stream>>>(hbuf, ln2g, ln2b, xn);
    gemm_bf16_kernel<2><<<gF, blk, 0, stream>>>(xn, wt_1[l], b1, nullptr, nullptr, midb, NTOK_, DFF_, D_);
    gemm_bf16_kernel<1><<<gD, blk, 0, stream>>>(midb, wt_2[l], b2, hbuf, hbuf, nullptr, NTOK_, D_, DFF_);
  }

  ln_bf16_kernel<<<NTOK_, blk, 0, stream>>>(hbuf, (const float*)d_in[68], (const float*)d_in[69], xn);
  gemm_bf16_kernel<0><<<dim3(VOCAB_ / 128, NTOK_ / 128), blk, 0, stream>>>(
      xn, wt_out, (const float*)d_in[71], nullptr, (float*)d_out, nullptr, NTOK_, VOCAB_, D_);
}
